// MambaBlock_47880295416210
// MI455X (gfx1250) — matrix operand order repaired and checked by reading, not compiled, not run on hardware
//
#include <hip/hip_runtime.h>
#include <hip/hip_bf16.h>
#include <stdint.h>

// ---------------------------------------------------------------------------
// Mamba block forward for MI455X (gfx1250, wave32, WMMA + async LDS staging).
// B=2, L=2048, D_MODEL=1024, D_INNER=2048, D_STATE=16, DT_RANK=64, D_CONV=4.
// ---------------------------------------------------------------------------

#define DMODEL 1024
#define DSTATE 16
#define DCONV  4
#define DINNER 2048
#define DTRANK 64
#define NBATCH 2
#define SEQLEN 2048
#define ROWS   (NBATCH * SEQLEN)      // 4096 (b*l rows)
#define XPROJN (DTRANK + 2 * DSTATE)  // 96

typedef __attribute__((ext_vector_type(16))) __bf16 v16bf;
typedef __attribute__((ext_vector_type(8)))  __bf16 v8bf;
typedef __attribute__((ext_vector_type(8)))  float  v8f;

#if __has_builtin(__builtin_amdgcn_global_load_async_to_lds_b128)
#define HAS_ASYNC_LDS 1
#else
#define HAS_ASYNC_LDS 0
#endif

// Pointer types matching the async builtin's signature:
//   (global v4i*, lds v4i*, int offset, int cpol)
typedef int v4i_g __attribute__((vector_size(16)));
typedef __attribute__((address_space(1))) v4i_g* as1_v4i_ptr;
typedef __attribute__((address_space(3))) v4i_g* as3_v4i_ptr;

// 16-byte copy global -> LDS. Async (ASYNCcnt-tracked) when available.
__device__ __forceinline__ void cp16(__bf16* lds_dst, const __bf16* gsrc) {
#if HAS_ASYNC_LDS
    __builtin_amdgcn_global_load_async_to_lds_b128(
        (as1_v4i_ptr)(uintptr_t)gsrc,
        (as3_v4i_ptr)(unsigned int)(uintptr_t)lds_dst,
        0, 0);
#else
    *(v8bf*)lds_dst = *(const v8bf*)gsrc;
#endif
}

__device__ __forceinline__ void async_fence_all() {
#if HAS_ASYNC_LDS
#if __has_builtin(__builtin_amdgcn_s_wait_asynccnt)
    __builtin_amdgcn_s_wait_asynccnt(0);
#else
    asm volatile("s_wait_asynccnt 0x0" ::: "memory");
#endif
#endif
}

// round-to-nearest-even f32 -> bf16 via bit ops (no fp conversion dependency)
__device__ __forceinline__ __bf16 f2bf(float f) {
    union { float f; unsigned int u; } v; v.f = f;
    unsigned int r = v.u + 0x7FFFu + ((v.u >> 16) & 1u);
    union { unsigned short s; __bf16 b; } o;
    o.s = (unsigned short)(r >> 16);
    return o.b;
}

__device__ __forceinline__ v16bf cat8(v8bf lo, v8bf hi) {
    return __builtin_shufflevector(lo, hi, 0, 1, 2, 3, 4, 5, 6, 7,
                                   8, 9, 10, 11, 12, 13, 14, 15);
}

__global__ __launch_bounds__(256)
void cvt_bf16_kernel(const float* __restrict__ src, __bf16* __restrict__ dst, int n) {
    int i = blockIdx.x * 256 + threadIdx.x;
    if (i < n) dst[i] = f2bf(src[i]);
}

// ---------------------------------------------------------------------------
// Shared epilogue. C layout: VGPR r -> row (r + 8*half), col l16 within tile.
//   MODE 0: plain store
//   MODE 1: += bias[col]; softplus       (delta = softplus(dt @ W^T + b))
//   MODE 2: += resid[row,col]            (out_proj + residual)
// ---------------------------------------------------------------------------
template <int MODE>
__device__ __forceinline__ void store_tile(float* __restrict__ C, const v8f& a,
                                           int row0, int col, int ldc, int half,
                                           const float* __restrict__ bias,
                                           const float* __restrict__ resid, int ldr) {
#pragma unroll
    for (int r = 0; r < 8; ++r) {
        const int row = row0 + r + 8 * half;
        float v = a[r];
        if (MODE == 1) {
            v += bias[col];
            v = (v > 20.0f) ? v : log1pf(expf(v));  // softplus
        }
        if (MODE == 2) {
            v += resid[(size_t)row * ldr + col];
        }
        C[(size_t)row * ldc + col] = v;
    }
}

// ---------------------------------------------------------------------------
// LDS-staged WMMA GEMM: C[M,N] = A[M,K] * Bw[N,K]^T  (bf16 in, f32 accumulate)
// Block tile 64(M) x 256(N): 8 waves as 2x4, wave tile 32x64, K-chunk 32.
// Double-buffered LDS filled with GLOBAL_LOAD_ASYNC_TO_LDS_B128; fragment
// reads come from LDS (ds_load_b128), rows padded +8 bf16 to skew banks.
// Requires M % 64 == 0, N % 256 == 0, K % 32 == 0.
// ---------------------------------------------------------------------------
#define LPAD 8
#define LSTR (32 + LPAD)   // LDS row stride in bf16 elements

template <int MODE>
__global__ __launch_bounds__(256)
void wmma_gemm_lds_kernel(const __bf16* __restrict__ A, const __bf16* __restrict__ Bw,
                          float* __restrict__ C, int M, int N, int K,
                          int lda, int ldb, int ldc,
                          const float* __restrict__ bias,
                          const float* __restrict__ resid, int ldr) {
    __shared__ __bf16 sA[2][64 * LSTR];    // 2 x 5.0 KB
    __shared__ __bf16 sB[2][256 * LSTR];   // 2 x 20.0 KB

    const int tid  = threadIdx.x;
    const int wave = tid >> 5;
    const int lane = tid & 31;
    const int wm   = wave >> 2;       // 0..1 (M)
    const int wn   = wave & 3;        // 0..3 (N)
    const int half = lane >> 4;       // 0..1
    const int l16  = lane & 15;       // 0..15

    const int tiles_n = N / 256;
    const int bm = blockIdx.x / tiles_n;
    const int bn = blockIdx.x % tiles_n;

    const __bf16* Ab = A  + (size_t)(bm * 64)  * lda;
    const __bf16* Bb = Bw + (size_t)(bn * 256) * ldb;

    // cooperative copy coordinates: 256 threads move A (64x32) and B (256x32)
    const int crow = tid >> 2;            // 0..63
    const int koff = (tid & 3) * 8;       // 0,8,16,24

    auto issue = [&](int buf, int k0) {
        cp16(&sA[buf][crow * LSTR + koff], Ab + (size_t)crow * lda + k0 + koff);
#pragma unroll
        for (int r = 0; r < 4; ++r)
            cp16(&sB[buf][(crow + r * 64) * LSTR + koff],
                 Bb + (size_t)(crow + r * 64) * ldb + k0 + koff);
    };

    v8f acc[2][4];
#pragma unroll
    for (int i = 0; i < 2; ++i)
#pragma unroll
        for (int j = 0; j < 4; ++j) acc[i][j] = (v8f)(0.0f);

    const int nk = K / 32;
    issue(0, 0);

    for (int kc = 0; kc < nk; ++kc) {
        const int buf = kc & 1;
        async_fence_all();       // my async fills of `buf` have landed
        __syncthreads();         // everyone's fills landed; prev reads of buf^1 done
        if (kc + 1 < nk) issue(buf ^ 1, (kc + 1) * 32);

        v16bf afrag[2];
#pragma unroll
        for (int i = 0; i < 2; ++i) {
            const __bf16* ap = &sA[buf][(wm * 32 + i * 16 + l16) * LSTR + half * 8];
            afrag[i] = cat8(*(const v8bf*)ap, *(const v8bf*)(ap + 16));
        }
#pragma unroll
        for (int j = 0; j < 4; ++j) {
            const __bf16* bp = &sB[buf][(wn * 64 + j * 16 + l16) * LSTR + half * 8];
            v16bf bfrag = cat8(*(const v8bf*)bp, *(const v8bf*)(bp + 16));
#pragma unroll
            for (int i = 0; i < 2; ++i) {
                acc[i][j] = __builtin_amdgcn_wmma_f32_16x16x32_bf16(
                    false, afrag[i], false, bfrag, (short)0, acc[i][j], false, false);
            }
        }
    }

#pragma unroll
    for (int i = 0; i < 2; ++i)
#pragma unroll
        for (int j = 0; j < 4; ++j)
            store_tile<MODE>(C, acc[i][j],
                             bm * 64 + wm * 32 + i * 16,
                             bn * 256 + wn * 64 + j * 16 + l16,
                             ldc, half, bias, resid, ldr);
}

// ---------------------------------------------------------------------------
// Direct-from-global WMMA GEMM (for small N, e.g. x_proj N=96).
// Wave tile (MT*16) x (NT*16).
// ---------------------------------------------------------------------------
template <int MT, int NT, int MODE>
__global__ __launch_bounds__(256)
void wmma_gemm_kernel(const __bf16* __restrict__ A, const __bf16* __restrict__ Bw,
                      float* __restrict__ C, int M, int N, int K,
                      int lda, int ldb, int ldc,
                      const float* __restrict__ bias,
                      const float* __restrict__ resid, int ldr) {
    const int wave    = (blockIdx.x * 256 + threadIdx.x) >> 5;
    const int lane    = threadIdx.x & 31;
    const int tiles_n = N / (NT * 16);
    const int tiles_m = M / (MT * 16);
    const int tm      = wave / tiles_n;
    const int tn      = wave % tiles_n;
    if (tm >= tiles_m) return;  // uniform per-wave exit; EXEC stays all-ones

    const int half = lane >> 4;
    const int l16  = lane & 15;

    v8f acc[MT][NT];
#pragma unroll
    for (int i = 0; i < MT; ++i)
#pragma unroll
        for (int j = 0; j < NT; ++j) acc[i][j] = (v8f)(0.0f);

    const __bf16* Abase = A  + (size_t)(tm * MT * 16) * lda;
    const __bf16* Bbase = Bw + (size_t)(tn * NT * 16) * ldb;

    for (int k0 = 0; k0 < K; k0 += 32) {
        if (k0 + 32 < K) {
            __builtin_prefetch(Abase + (size_t)l16 * lda + k0 + 32, 0, 3);
            __builtin_prefetch(Bbase + (size_t)l16 * ldb + k0 + 32, 0, 3);
        }
        v16bf afrag[MT];
#pragma unroll
        for (int i = 0; i < MT; ++i) {
            const __bf16* ap = Abase + (size_t)(i * 16 + l16) * lda + k0 + half * 8;
            afrag[i] = cat8(*(const v8bf*)(ap), *(const v8bf*)(ap + 16));
        }
#pragma unroll
        for (int j = 0; j < NT; ++j) {
            const __bf16* bp = Bbase + (size_t)(j * 16 + l16) * ldb + k0 + half * 8;
            v16bf bfrag = cat8(*(const v8bf*)bp, *(const v8bf*)(bp + 16));
#pragma unroll
            for (int i = 0; i < MT; ++i) {
                acc[i][j] = __builtin_amdgcn_wmma_f32_16x16x32_bf16(
                    false, afrag[i], false, bfrag, (short)0, acc[i][j], false, false);
            }
        }
    }

#pragma unroll
    for (int i = 0; i < MT; ++i)
#pragma unroll
        for (int j = 0; j < NT; ++j)
            store_tile<MODE>(C, acc[i][j],
                             tm * MT * 16 + i * 16,
                             tn * NT * 16 + j * 16 + l16,
                             ldc, half, bias, resid, ldr);
}

// ---------------------------------------------------------------------------
// Causal depthwise conv1d (+bias) + SiLU over xi = xz[:, :DINNER]
// ---------------------------------------------------------------------------
__global__ __launch_bounds__(256)
void conv_silu_kernel(const float* __restrict__ xz, const float* __restrict__ cw,
                      const float* __restrict__ cb, float* __restrict__ u,
                      __bf16* __restrict__ u_bf) {
    const long long idx = (long long)blockIdx.x * 256 + threadIdx.x;
    if (idx >= (long long)ROWS * DINNER) return;
    const int d  = (int)(idx % DINNER);
    const int bl = (int)(idx / DINNER);
    const int l  = bl % SEQLEN;
    const int b  = bl / SEQLEN;

    float acc = cb[d];
#pragma unroll
    for (int k = 0; k < DCONV; ++k) {
        const int ls = l - (DCONV - 1) + k;
        if (ls >= 0)
            acc += xz[((size_t)(b * SEQLEN + ls)) * (2 * DINNER) + d] * cw[d * DCONV + k];
    }
    const float s = acc / (1.0f + expf(-acc));  // silu
    u[(size_t)bl * DINNER + d]    = s;
    u_bf[(size_t)bl * DINNER + d] = f2bf(s);
}

// ---------------------------------------------------------------------------
// Selective scan. One thread per (b, d) channel; 16-state recurrence over L.
// B_t / C_t broadcast via LDS. Fused: D-skip + SiLU(z) gate, bf16 output.
// ---------------------------------------------------------------------------
__global__ __launch_bounds__(256)
void scan_kernel(const float* __restrict__ delta, const float* __restrict__ u,
                 const float* __restrict__ x_dbl, const float* __restrict__ xz,
                 const float* __restrict__ A_log, const float* __restrict__ Dp,
                 __bf16* __restrict__ y_bf) {
    const int b = blockIdx.x / (DINNER / 256);
    const int d = (blockIdx.x % (DINNER / 256)) * 256 + threadIdx.x;

    float Aneg[DSTATE], h[DSTATE];
#pragma unroll
    for (int n = 0; n < DSTATE; ++n) {
        Aneg[n] = -expf(A_log[(size_t)d * DSTATE + n]);
        h[n] = 0.0f;
    }
    const float Dd = Dp[d];

    __shared__ float sB[DSTATE];
    __shared__ float sC[DSTATE];

    for (int l = 0; l < SEQLEN; ++l) {
        const size_t row = (size_t)b * SEQLEN + l;
        if (threadIdx.x < DSTATE)
            sB[threadIdx.x] = x_dbl[row * XPROJN + DTRANK + threadIdx.x];
        else if (threadIdx.x < 2 * DSTATE)
            sC[threadIdx.x - DSTATE] = x_dbl[row * XPROJN + DTRANK + DSTATE + (threadIdx.x - DSTATE)];
        __syncthreads();

        const float dt = delta[row * DINNER + d];
        const float ut = u[row * DINNER + d];
        const float du = dt * ut;
        float y = 0.0f;
#pragma unroll
        for (int n = 0; n < DSTATE; ++n) {
            h[n] = expf(dt * Aneg[n]) * h[n] + du * sB[n];
            y += h[n] * sC[n];
        }
        y += ut * Dd;
        const float z = xz[row * (2 * DINNER) + DINNER + d];
        y *= z / (1.0f + expf(-z));  // * silu(z)
        y_bf[row * DINNER + d] = f2bf(y);
        __syncthreads();
    }
}

// ---------------------------------------------------------------------------
// Row LayerNorm over D_MODEL=1024. One block per row.
// ---------------------------------------------------------------------------
__global__ __launch_bounds__(256)
void layernorm_kernel(const float* __restrict__ h, const float* __restrict__ w,
                      const float* __restrict__ bta, float* __restrict__ out) {
    const int row = blockIdx.x;
    const float* hr = h + (size_t)row * DMODEL;
    float s = 0.0f, s2 = 0.0f;
    for (int c = threadIdx.x; c < DMODEL; c += 256) {
        const float v = hr[c];
        s += v; s2 += v * v;
    }
    __shared__ float rs[256], rs2[256];
    rs[threadIdx.x] = s; rs2[threadIdx.x] = s2;
    __syncthreads();
    for (int st = 128; st > 0; st >>= 1) {
        if (threadIdx.x < st) {
            rs[threadIdx.x]  += rs[threadIdx.x + st];
            rs2[threadIdx.x] += rs2[threadIdx.x + st];
        }
        __syncthreads();
    }
    const float mu  = rs[0] * (1.0f / DMODEL);
    const float var = rs2[0] * (1.0f / DMODEL) - mu * mu;
    const float inv = rsqrtf(var + 1e-5f);
    for (int c = threadIdx.x; c < DMODEL; c += 256)
        out[(size_t)row * DMODEL + c] = (hr[c] - mu) * inv * w[c] + bta[c];
}

// ---------------------------------------------------------------------------
// Host launcher
// ---------------------------------------------------------------------------
extern "C" void kernel_launch(void* const* d_in, const int* in_sizes, int n_in,
                              void* d_out, int out_size, void* d_ws, size_t ws_size,
                              hipStream_t stream) {
    const float* x        = (const float*)d_in[0];   // [2,2048,1024]
    const float* in_w     = (const float*)d_in[1];   // [4096,1024]
    const float* conv_w   = (const float*)d_in[2];   // [2048,1,4]
    const float* conv_b   = (const float*)d_in[3];   // [2048]
    const float* xproj_w  = (const float*)d_in[4];   // [96,2048]
    const float* dtproj_w = (const float*)d_in[5];   // [2048,64]
    const float* dtproj_b = (const float*)d_in[6];   // [2048]
    const float* A_log    = (const float*)d_in[7];   // [2048,16]
    const float* Dp       = (const float*)d_in[8];   // [2048]
    const float* out_w    = (const float*)d_in[9];   // [1024,2048]
    const float* ln_w     = (const float*)d_in[10];  // [1024]
    const float* ln_b     = (const float*)d_in[11];  // [1024]

    char* ws = (char*)d_ws;
    size_t off = 0;
    auto carve = [&](size_t bytes) {
        size_t r = off;
        off += (bytes + 255) & ~(size_t)255;
        return r;
    };
    float*  xz       = (float*)(ws + carve((size_t)ROWS * 2 * DINNER * 4));   // 64 MB
    float*  u        = (float*)(ws + carve((size_t)ROWS * DINNER * 4));       // 32 MB
    float*  x_dbl    = (float*)(ws + carve((size_t)ROWS * XPROJN * 4));       // 1.5 MB
    float*  delta    = (float*)(ws + carve((size_t)ROWS * DINNER * 4));       // 32 MB
    float*  h_pre    = (float*)(ws + carve((size_t)ROWS * DMODEL * 4));       // 16 MB
    __bf16* x_bf     = (__bf16*)(ws + carve((size_t)ROWS * DMODEL * 2));
    __bf16* w_in_bf  = (__bf16*)(ws + carve((size_t)(2 * DINNER) * DMODEL * 2));
    __bf16* w_xp_bf  = (__bf16*)(ws + carve((size_t)XPROJN * DINNER * 2));
    __bf16* w_dt_bf  = (__bf16*)(ws + carve((size_t)DINNER * DTRANK * 2));
    __bf16* w_out_bf = (__bf16*)(ws + carve((size_t)DMODEL * DINNER * 2));
    __bf16* u_bf     = (__bf16*)(ws + carve((size_t)ROWS * DINNER * 2));
    __bf16* xdbl_bf  = (__bf16*)(ws + carve((size_t)ROWS * XPROJN * 2));
    __bf16* y_bf     = (__bf16*)(ws + carve((size_t)ROWS * DINNER * 2));
    (void)ws_size; (void)n_in; (void)in_sizes; (void)out_size;

    auto cvt = [&](const float* s, __bf16* dptr, int n) {
        cvt_bf16_kernel<<<(n + 255) / 256, 256, 0, stream>>>(s, dptr, n);
    };
    cvt(x,        x_bf,     ROWS * DMODEL);
    cvt(in_w,     w_in_bf,  2 * DINNER * DMODEL);
    cvt(xproj_w,  w_xp_bf,  XPROJN * DINNER);
    cvt(dtproj_w, w_dt_bf,  DINNER * DTRANK);
    cvt(out_w,    w_out_bf, DMODEL * DINNER);

    auto lds_blocks = [](int M, int N) { return (M / 64) * (N / 256); };

    // 1) xz = x @ in_proj_w^T      [4096 x 4096], K=1024
    wmma_gemm_lds_kernel<0><<<lds_blocks(ROWS, 2 * DINNER), 256, 0, stream>>>(
        x_bf, w_in_bf, xz, ROWS, 2 * DINNER, DMODEL, DMODEL, DMODEL, 2 * DINNER,
        nullptr, nullptr, 0);

    // 2) u = silu(causal_dwconv(xi) + conv_b)
    conv_silu_kernel<<<(ROWS * DINNER + 255) / 256, 256, 0, stream>>>(
        xz, conv_w, conv_b, u, u_bf);

    // 3) x_dbl = u @ x_proj_w^T    [4096 x 96], K=2048 (small N: direct kernel)
    {
        const int waves = (ROWS / 32) * (XPROJN / 48);
        wmma_gemm_kernel<2, 3, 0><<<(waves + 7) / 8, 256, 0, stream>>>(
            u_bf, w_xp_bf, x_dbl, ROWS, XPROJN, DINNER, DINNER, DINNER, XPROJN,
            nullptr, nullptr, 0);
    }

    // 4) delta = softplus(dt @ dt_proj_w^T + dt_proj_b)   [4096 x 2048], K=64
    cvt(x_dbl, xdbl_bf, ROWS * XPROJN);
    wmma_gemm_lds_kernel<1><<<lds_blocks(ROWS, DINNER), 256, 0, stream>>>(
        xdbl_bf, w_dt_bf, delta, ROWS, DINNER, DTRANK, XPROJN, DTRANK, DINNER,
        dtproj_b, nullptr, 0);

    // 5) selective scan + D-skip + SiLU(z) gate -> y (bf16)
    scan_kernel<<<NBATCH * (DINNER / 256), 256, 0, stream>>>(
        delta, u, x_dbl, xz, A_log, Dp, y_bf);

    // 6) h_pre = y @ out_proj_w^T + x (residual)   [4096 x 1024], K=2048
    wmma_gemm_lds_kernel<2><<<lds_blocks(ROWS, DMODEL), 256, 0, stream>>>(
        y_bf, w_out_bf, h_pre, ROWS, DMODEL, DINNER, DINNER, DINNER, DMODEL,
        nullptr, x, DMODEL);

    // 7) LayerNorm -> d_out
    layernorm_kernel<<<ROWS, 256, 0, stream>>>(h_pre, ln_w, ln_b, (float*)d_out);
}
